// Model_79096117723518
// MI455X (gfx1250) — matrix operand order repaired and checked by reading, not compiled, not run on hardware
//
#include <hip/hip_runtime.h>
#include <hip/hip_bf16.h>
#include <math.h>

// ---------------------------------------------------------------------------
// CDNA5 (gfx1250): CNN -> ConvLSTM2D -> dense head, all GEMM work on
// v_wmma_f32_16x16x32_f16.  BN folded into weights; weights pre-shuffled into
// WMMA fragment order so every operand is fetched as 2x global_load_b128.
// LSTM gates fused into the recurrent GEMM via an LDS tile exchange.
// ---------------------------------------------------------------------------

typedef __attribute__((ext_vector_type(16))) _Float16     v16h;
typedef __attribute__((ext_vector_type(8)))  float        v8f;
typedef __attribute__((ext_vector_type(4)))  unsigned int v4u;
typedef __attribute__((ext_vector_type(4)))  float        v4f;

union Frag { v16h h; v4u u[2]; };

#define EPSBN 1e-3f
#define BATCH 32
#define TSTEP 12
#define P144  144
#define M3    4608
#define PH    294912        // M3*64, one timestep of h

__device__ __forceinline__ float hsig(float x) {
  return fminf(fmaxf(0.2f * x + 0.5f, 0.f), 1.f);
}
__device__ __forceinline__ v8f wmma16(v16h a, v16h b, v8f c) {
  return __builtin_amdgcn_wmma_f32_16x16x32_f16(false, a, false, b,
                                                (short)0, c, false, false);
}
// B fragment: pre-shuffled weights, 32 bytes per lane per k-tile.
__device__ __forceinline__ v16h loadB(const _Float16* __restrict__ frag,
                                      long tileIdx, int lane) {
  Frag fb;
  const v4u* p = (const v4u*)(frag + (tileIdx * 32 + lane) * 16);
  fb.u[0] = p[0];
  fb.u[1] = p[1];
  return fb.h;
}

// ---------------------------- prep kernels ---------------------------------

// Shuffle weight [K][N] (f32) into WMMA B-fragment order (f16), folding an
// optional BN scale s[c], c = k % cmod, and zero-padding k >= K.
__global__ __launch_bounds__(256) void k_make_frag(
    const float* __restrict__ w, const float* __restrict__ g,
    const float* __restrict__ vv, _Float16* __restrict__ frag,
    int K, int N, int KT, int cmod, int total) {
  int idx = blockIdx.x * blockDim.x + threadIdx.x;
  if (idx >= total) return;
  int i    = idx & 15;
  int lane = (idx >> 4) & 31;
  int rest = idx >> 9;
  int kt   = rest % KT;
  int nt   = rest / KT;
  int k = kt * 32 + (i >> 3) * 16 + (lane >> 4) * 8 + (i & 7);
  int n = nt * 16 + (lane & 15);
  float val = 0.f;
  if (k < K) {
    float s = 1.f;
    if (cmod > 0) {
      int c = k % cmod;
      s = g[c] / sqrtf(vv[c] + EPSBN);
    }
    val = w[k * N + n] * s;
  }
  frag[idx] = (_Float16)val;
}

// bf[o] = bias0[o] + sum_k w[k*N+o] * (b[c] - m[c]*s[c]),  c = k % cmod
__global__ __launch_bounds__(256) void k_fold_bias(
    const float* __restrict__ w, const float* __restrict__ bias0,
    const float* __restrict__ g, const float* __restrict__ bb,
    const float* __restrict__ mm, const float* __restrict__ vv,
    float* __restrict__ bf, int K, int N, int cmod) {
  int o = blockIdx.x * blockDim.x + threadIdx.x;
  if (o >= N) return;
  float acc = bias0[o];
  for (int k = 0; k < K; ++k) {
    int c = k % cmod;
    float s = g[c] / sqrtf(vv[c] + EPSBN);
    acc += w[k * N + o] * (bb[c] - mm[c] * s);
  }
  bf[o] = acc;
}

__global__ __launch_bounds__(256) void k_zero_h(_Float16* p, int n) {
  int i = blockIdx.x * blockDim.x + threadIdx.x;
  if (i < n) p[i] = (_Float16)0.f;
}
__global__ __launch_bounds__(256) void k_zero_f(float* p, int n) {
  int i = blockIdx.x * blockDim.x + threadIdx.x;
  if (i < n) p[i] = 0.f;
}

// ------------------------- conv1: 4x4 s2 VALID, 4->16 ----------------------
// M = 369024, N = 16, K = 64 (2 k-tiles).  A runs of 8 k = 8 contiguous f32
// (2 adjacent pixels x 4 channels) -> two 16B f32 loads, cvt to f16.
__global__ __launch_bounds__(128) void k_conv1(
    const float* __restrict__ x, const _Float16* __restrict__ wfrag,
    const float* __restrict__ bf, _Float16* __restrict__ out) {
  int wave = blockIdx.x * 4 + (threadIdx.x >> 5);
  int lane = threadIdx.x & 31, rowm = lane & 15, hi = lane >> 4;
  int mbase = wave * 16;
  int m  = mbase + rowm;
  int bt = m / 961, rem = m - bt * 961;
  int oy = rem / 31, ox = rem - oy * 31;
  long ibase = (((long)bt * 64 + oy * 2) * 64 + ox * 2) * 4;
  float bias = bf[rowm];
  v8f acc;
#pragma unroll
  for (int i = 0; i < 8; ++i) acc[i] = bias;
#pragma unroll
  for (int kt = 0; kt < 2; ++kt) {
    v16h a;
#pragma unroll
    for (int r = 0; r < 2; ++r) {
      int k0 = kt * 32 + r * 16 + hi * 8;     // k = kh*16 + kw*4 + c
      int kh = k0 >> 4, kw0 = (k0 >> 2) & 3;
      const float* px = x + ibase + (kh * 64 + kw0) * 4;
      v4f f0 = *(const v4f*)px;
      v4f f1 = *(const v4f*)(px + 4);
#pragma unroll
      for (int j = 0; j < 4; ++j) {
        a[r * 8 + j]     = (_Float16)f0[j];
        a[r * 8 + 4 + j] = (_Float16)f1[j];
      }
    }
    acc = wmma16(a, loadB(wfrag, kt, lane), acc);
  }
#pragma unroll
  for (int i = 0; i < 8; ++i) {
    int mm = mbase + hi * 8 + i;
    float v = acc[i];
    out[(long)mm * 16 + rowm] = (_Float16)(v > 0.f ? v : 0.f);
  }
}

// ------------------------- conv2: 3x3 s2 VALID, 16->32 ---------------------
// M = 86400, N = 32 (2 tiles), K = 144 -> 5 k-tiles (zeros baked into frags).
__global__ __launch_bounds__(128) void k_conv2(
    const _Float16* __restrict__ a1, const _Float16* __restrict__ wfrag,
    const float* __restrict__ bf, _Float16* __restrict__ out) {
  int wave = blockIdx.x * 4 + (threadIdx.x >> 5);
  int mt = wave >> 1, nt = wave & 1;
  int lane = threadIdx.x & 31, rowm = lane & 15, hi = lane >> 4;
  int mbase = mt * 16;
  int m  = mbase + rowm;
  int bt = m / 225, rem = m - bt * 225;
  int oy = rem / 15, ox = rem - oy * 15;
  int n  = nt * 16 + rowm;
  long ibase = (((long)bt * 31 + oy * 2) * 31 + ox * 2) * 16;
  float bias = bf[n];
  v8f acc;
#pragma unroll
  for (int i = 0; i < 8; ++i) acc[i] = bias;
#pragma unroll
  for (int kt = 0; kt < 5; ++kt) {
    Frag fa;
#pragma unroll
    for (int r = 0; r < 2; ++r) {
      int k0 = kt * 32 + r * 16 + hi * 8;     // k = (kh*3+kw)*16 + c
      if (k0 < 144) {
        int kk = k0 >> 4, kh = kk / 3, kw = kk - kh * 3, c0 = k0 & 15;
        fa.u[r] = *(const v4u*)(a1 + ibase + (kh * 31 + kw) * 16 + c0);
      } else {
        v4u z = {0u, 0u, 0u, 0u};
        fa.u[r] = z;
      }
    }
    acc = wmma16(fa.h, loadB(wfrag, (long)nt * 5 + kt, lane), acc);
  }
#pragma unroll
  for (int i = 0; i < 8; ++i) {
    int mm = mbase + hi * 8 + i;
    float v = acc[i];
    out[(long)mm * 32 + n] = (_Float16)(v > 0.f ? v : 0.f);
  }
}

// ----- fused ConvLSTM step:  z GEMM (48 WMMA) + LDS exchange + gates -------
// Block = 4 waves.  Block (mt, q) computes m-tile mt for channels 16q..16q+15
// of all four gates: wave w handles nt = q + 4*w  (gate w).  z tiles meet in
// LDS; gate math writes c (f32) and h (f16) directly.  No z round-trip.
__global__ __launch_bounds__(128) void k_lstm_step(
    const _Float16* __restrict__ a2, const _Float16* __restrict__ hprev,
    const _Float16* __restrict__ wxfrag, const _Float16* __restrict__ whfrag,
    const float* __restrict__ bzf, float* __restrict__ cbuf,
    _Float16* __restrict__ hs_t, int t) {
  __shared__ float zls[4][16][17];            // +1 pad: conflict-free columns
  int w    = threadIdx.x >> 5;                // wave id = gate id
  int mt   = blockIdx.x >> 2;
  int q    = blockIdx.x & 3;                  // channel quarter within gate
  int nt   = q + 4 * w;
  int lane = threadIdx.x & 31, rowm = lane & 15, hi = lane >> 4;
  int mbase = mt * 16;
  int m  = mbase + rowm;
  int b_ = m / P144, p = m - b_ * P144;
  int y  = p / 12, x0 = p - y * 12;
  int n  = nt * 16 + rowm;
  long xbase = ((((long)b_ * TSTEP + t) * 15 + y) * 15 + x0) * 32;
  long hbase = (long)b_ * P144 * 64;
  float bias = bzf[n];
  v8f acc;
#pragma unroll
  for (int i = 0; i < 8; ++i) acc[i] = bias;

  // x-conv: K = 4*4*32 = 512
#pragma unroll
  for (int kt = 0; kt < 16; ++kt) {
    Frag fa;
#pragma unroll
    for (int r = 0; r < 2; ++r) {
      int k0 = kt * 32 + r * 16 + hi * 8;     // k = (kh*4+kw)*32 + c
      int kk = k0 >> 5, kh = kk >> 2, kw = kk & 3, c0 = k0 & 31;
      fa.u[r] = *(const v4u*)(a2 + xbase + (kh * 15 + kw) * 32 + c0);
    }
    acc = wmma16(fa.h, loadB(wxfrag, (long)nt * 16 + kt, lane), acc);
  }
  // h-conv: K = 4*4*64 = 1024, SAME (offset -1, zero pad)
#pragma unroll
  for (int kt = 0; kt < 32; ++kt) {
    Frag fa;
#pragma unroll
    for (int r = 0; r < 2; ++r) {
      int k0 = kt * 32 + r * 16 + hi * 8;     // k = (kh*4+kw)*64 + c
      int kk = k0 >> 6, kh = kk >> 2, kw = kk & 3, c0 = k0 & 63;
      int yy = y - 1 + kh, xw = x0 - 1 + kw;
      if ((unsigned)yy < 12u && (unsigned)xw < 12u) {
        fa.u[r] = *(const v4u*)(hprev + hbase + (yy * 12 + xw) * 64 + c0);
      } else {
        v4u zv = {0u, 0u, 0u, 0u};
        fa.u[r] = zv;
      }
    }
    acc = wmma16(fa.h, loadB(whfrag, (long)nt * 32 + kt, lane), acc);
  }
  // park this gate's 16x16 z tile in LDS
#pragma unroll
  for (int i = 0; i < 8; ++i) zls[w][hi * 8 + i][rowm] = acc[i];
  __syncthreads();

  // gate math: 256 elements (16 m-rows x 16 channels), 2 per thread
#pragma unroll
  for (int e = 0; e < 2; ++e) {
    int idx  = threadIdx.x * 2 + e;
    int mrow = idx >> 4, ch = idx & 15;
    float zi = zls[0][mrow][ch];
    float zf = zls[1][mrow][ch];
    float zg = zls[2][mrow][ch];
    float zo = zls[3][mrow][ch];
    long cidx = (long)(mbase + mrow) * 64 + q * 16 + ch;
    float cn = hsig(zf) * cbuf[cidx] + hsig(zi) * tanhf(zg);
    float hn = hsig(zo) * tanhf(cn);
    cbuf[cidx] = cn;
    hs_t[cidx] = (_Float16)hn;
  }
}

// -------- 1x1 conv over time-stacked h: M=4608, N=64, K=768 ----------------
__global__ __launch_bounds__(128) void k_comp(
    const _Float16* __restrict__ hs, const _Float16* __restrict__ wofrag,
    const float* __restrict__ bo, _Float16* __restrict__ flat) {
  int wave = blockIdx.x * 4 + (threadIdx.x >> 5);
  int mt = wave >> 2, nt = wave & 3;
  int lane = threadIdx.x & 31, rowm = lane & 15, hi = lane >> 4;
  int mbase = mt * 16;
  int m  = mbase + rowm;
  int b_ = m / P144, p = m - b_ * P144;   // 144 % 16 == 0
  int n  = nt * 16 + rowm;
  float bias = bo[n];
  v8f acc;
#pragma unroll
  for (int i = 0; i < 8; ++i) acc[i] = bias;
#pragma unroll
  for (int kt = 0; kt < 24; ++kt) {
    Frag fa;
#pragma unroll
    for (int r = 0; r < 2; ++r) {
      int k0 = kt * 32 + r * 16 + hi * 8;     // k = t*64 + c
      int tt = k0 >> 6, c0 = k0 & 63;
      fa.u[r] = *(const v4u*)(hs + ((long)tt * BATCH + b_) * P144 * 64 +
                              (long)p * 64 + c0);
    }
    acc = wmma16(fa.h, loadB(wofrag, (long)nt * 24 + kt, lane), acc);
  }
#pragma unroll
  for (int i = 0; i < 8; ++i) {
    int mm = mbase + hi * 8 + i;
    int b2 = mm / P144, p2 = mm - b2 * P144;
    float v = acc[i];
    flat[(long)b2 * 9216 + p2 * 64 + n] = (_Float16)(v > 0.f ? v : 0.f);
  }
}

// ----------------- dense1: [32,9216] @ [9216,128] + relu -------------------
__global__ __launch_bounds__(128) void k_dense1(
    const _Float16* __restrict__ flat, const _Float16* __restrict__ wfrag,
    const float* __restrict__ bd, float* __restrict__ hdn) {
  int wave = blockIdx.x * 4 + (threadIdx.x >> 5);
  int mt = wave >> 3, nt = wave & 7;
  int lane = threadIdx.x & 31, rowm = lane & 15, hi = lane >> 4;
  int mbase = mt * 16;
  int m = mbase + rowm;
  int n = nt * 16 + rowm;
  float bias = bd[n];
  v8f acc;
#pragma unroll
  for (int i = 0; i < 8; ++i) acc[i] = bias;
#pragma unroll 4
  for (int kt = 0; kt < 288; ++kt) {
    Frag fa;
#pragma unroll
    for (int r = 0; r < 2; ++r) {
      int k0 = kt * 32 + r * 16 + hi * 8;
      fa.u[r] = *(const v4u*)(flat + (long)m * 9216 + k0);
    }
    acc = wmma16(fa.h, loadB(wfrag, (long)nt * 288 + kt, lane), acc);
  }
#pragma unroll
  for (int i = 0; i < 8; ++i) {
    int mm = mbase + hi * 8 + i;
    float v = acc[i];
    hdn[(long)mm * 128 + n] = v > 0.f ? v : 0.f;
  }
}

// ----------------- dense2 (128->2) + softmax, one thread/batch -------------
__global__ __launch_bounds__(32) void k_dense2_softmax(
    const float* __restrict__ hdn, const float* __restrict__ w2,
    const float* __restrict__ b2, float* __restrict__ out) {
  int b_ = threadIdx.x;
  if (b_ >= BATCH) return;
  float l0 = b2[0], l1 = b2[1];
  for (int k = 0; k < 128; ++k) {
    float h = hdn[(long)b_ * 128 + k];
    l0 += h * w2[k * 2 + 0];
    l1 += h * w2[k * 2 + 1];
  }
  float mx = fmaxf(l0, l1);
  float e0 = expf(l0 - mx), e1 = expf(l1 - mx);
  float s = e0 + e1;
  out[b_ * 2 + 0] = e0 / s;
  out[b_ * 2 + 1] = e1 / s;
}

// ---------------------------------------------------------------------------
extern "C" void kernel_launch(void* const* d_in, const int* in_sizes, int n_in,
                              void* d_out, int out_size, void* d_ws, size_t ws_size,
                              hipStream_t stream) {
  const float* xin   = (const float*)d_in[0];
  const float* bn0g  = (const float*)d_in[1];
  const float* bn0b  = (const float*)d_in[2];
  const float* bn0m  = (const float*)d_in[3];
  const float* bn0v  = (const float*)d_in[4];
  const float* w1    = (const float*)d_in[5];
  const float* b1    = (const float*)d_in[6];
  const float* bn1g  = (const float*)d_in[7];
  const float* bn1b  = (const float*)d_in[8];
  const float* bn1m  = (const float*)d_in[9];
  const float* bn1v  = (const float*)d_in[10];
  const float* w2    = (const float*)d_in[11];
  const float* b2    = (const float*)d_in[12];
  const float* bn2g  = (const float*)d_in[13];
  const float* bn2b  = (const float*)d_in[14];
  const float* bn2m  = (const float*)d_in[15];
  const float* bn2v  = (const float*)d_in[16];
  const float* wx    = (const float*)d_in[17];
  const float* wh    = (const float*)d_in[18];
  const float* blstm = (const float*)d_in[19];
  const float* wo    = (const float*)d_in[20];
  const float* bo    = (const float*)d_in[21];
  const float* wd1   = (const float*)d_in[22];
  const float* bd1   = (const float*)d_in[23];
  const float* wd2   = (const float*)d_in[24];
  const float* bd2   = (const float*)d_in[25];
  float* outp = (float*)d_out;

  char* ws = (char*)d_ws;
  size_t off = 0;
  auto alloc = [&](size_t bytes) -> void* {
    off = (off + 255) & ~(size_t)255;
    void* p = ws + off;
    off += bytes;
    return p;
  };
  // fragment sizes: (N/16) * KT * 512 halves
  _Float16* w1f  = (_Float16*)alloc((size_t)1 * 2 * 512 * 2);     // 1024
  _Float16* w2f  = (_Float16*)alloc((size_t)2 * 5 * 512 * 2);     // 5120
  _Float16* wxf  = (_Float16*)alloc((size_t)16 * 16 * 512 * 2);   // 131072
  _Float16* whf  = (_Float16*)alloc((size_t)16 * 32 * 512 * 2);   // 262144
  _Float16* wof  = (_Float16*)alloc((size_t)4 * 24 * 512 * 2);    // 49152
  _Float16* wd1f = (_Float16*)alloc((size_t)8 * 288 * 512 * 2);   // 1179648
  float*    b1f  = (float*)alloc(16 * 4);
  float*    b2f  = (float*)alloc(32 * 4);
  float*    bzf  = (float*)alloc(256 * 4);
  _Float16* a1   = (_Float16*)alloc((size_t)5904384 * 2);  // [384,31,31,16]
  _Float16* a2   = (_Float16*)alloc((size_t)2764800 * 2);  // [384,15,15,32]
  _Float16* hz   = (_Float16*)alloc((size_t)PH * 2);
  float*    cbuf = (float*)alloc((size_t)PH * 4);
  _Float16* hs   = (_Float16*)alloc((size_t)TSTEP * PH * 2);
  _Float16* flat = (_Float16*)alloc((size_t)BATCH * 9216 * 2);
  float*    hdn  = (float*)alloc((size_t)BATCH * 128 * 4);
  (void)ws_size; (void)in_sizes; (void)n_in; (void)out_size;

  // ---- weight fragment shuffles (BN folded where applicable)
  k_make_frag<<<(1024 + 255) / 256, 256, 0, stream>>>(w1, bn0g, bn0v, w1f, 64, 16, 2, 4, 1024);
  k_make_frag<<<(5120 + 255) / 256, 256, 0, stream>>>(w2, bn1g, bn1v, w2f, 144, 32, 5, 16, 5120);
  k_make_frag<<<(131072 + 255) / 256, 256, 0, stream>>>(wx, bn2g, bn2v, wxf, 512, 256, 16, 32, 131072);
  k_make_frag<<<(262144 + 255) / 256, 256, 0, stream>>>(wh, nullptr, nullptr, whf, 1024, 256, 32, 0, 262144);
  k_make_frag<<<(49152 + 255) / 256, 256, 0, stream>>>(wo, nullptr, nullptr, wof, 768, 64, 24, 0, 49152);
  k_make_frag<<<(1179648 + 255) / 256, 256, 0, stream>>>(wd1, nullptr, nullptr, wd1f, 9216, 128, 288, 0, 1179648);
  k_fold_bias<<<1, 16, 0, stream>>>(w1, b1, bn0g, bn0b, bn0m, bn0v, b1f, 64, 16, 4);
  k_fold_bias<<<1, 32, 0, stream>>>(w2, b2, bn1g, bn1b, bn1m, bn1v, b2f, 144, 32, 16);
  k_fold_bias<<<1, 256, 0, stream>>>(wx, blstm, bn2g, bn2b, bn2m, bn2v, bzf, 512, 256, 32);
  k_zero_h<<<(PH + 255) / 256, 256, 0, stream>>>(hz, PH);
  k_zero_f<<<(PH + 255) / 256, 256, 0, stream>>>(cbuf, PH);

  // ---- CNN front-end
  k_conv1<<<23064 / 4, 128, 0, stream>>>(xin, w1f, b1f, a1);   // 369024/16 waves
  k_conv2<<<10800 / 4, 128, 0, stream>>>(a1, w2f, b2f, a2);

  // ---- ConvLSTM recurrence (fused GEMM + gates, one kernel per step)
  for (int t = 0; t < TSTEP; ++t) {
    const _Float16* hprev = (t == 0) ? hz : (hs + (size_t)(t - 1) * PH);
    k_lstm_step<<<1152, 128, 0, stream>>>(a2, hprev, wxf, whf, bzf, cbuf,
                                          hs + (size_t)t * PH, t);
  }

  // ---- head
  k_comp<<<1152 / 4, 128, 0, stream>>>(hs, wof, bo, flat);
  k_dense1<<<16 / 4, 128, 0, stream>>>(flat, wd1f, bd1, hdn);
  k_dense2_softmax<<<1, 32, 0, stream>>>(hdn, wd2, bd2, outp);
}
